// get_model_4973572129195
// MI455X (gfx1250) — matrix operand order repaired and checked by reading, not compiled, not run on hardware
//
#include <hip/hip_runtime.h>
#include <hip/hip_bf16.h>

// ---------------------------------------------------------------------------
// Point2Skeleton-style PointNet++ forward for MI455X (gfx1250, wave32).
// All dense layers run through v_wmma_f32_16x16x32_f16 with LDS double
// buffering. Weights are pre-swizzled into K32-chunked N-major tiles so that
// both global->LDS staging and per-lane fragment reads are contiguous
// (ds_load_b128), eliminating scalar ds_load_u16 gathers in the hot loop.
// ---------------------------------------------------------------------------

typedef _Float16 h16;
typedef __attribute__((ext_vector_type(16))) _Float16 v16h;
typedef __attribute__((ext_vector_type(8)))  float    v8f;

struct alignas(16) V4 { unsigned int w[4]; };   // 16 bytes = 8 halves

#define BN_SCALE_F 0.9999950000374996f          // 1/sqrt(1+1e-5)
#define NBATCH 4

static inline int al32(int x) { return (x + 31) & ~31; }
static inline int al64(int x) { return (x + 63) & ~63; }

// ---------------------------------------------------------------------------
// 0) split input [B][6][N] into xyz [B][N][3] and feat [B][N][3]
// ---------------------------------------------------------------------------
__global__ void k_transpose_in(const float* __restrict__ xyz6,
                               float* __restrict__ lxyz,
                               float* __restrict__ feat, int n) {
  int i = blockIdx.x * 256 + threadIdx.x;
  if (i >= NBATCH * n) return;
  int b = i / n, p = i % n;
  const float* src = xyz6 + (size_t)b * 6 * n;
  for (int c = 0; c < 3; ++c) {
    lxyz[(size_t)i * 3 + c] = src[(size_t)c * n + p];
    feat[(size_t)i * 3 + c] = src[(size_t)(3 + c) * n + p];
  }
}

// ---------------------------------------------------------------------------
// 1) farthest point sampling (one block per batch, matches jax.lax.scan body)
// ---------------------------------------------------------------------------
__global__ void k_fps(const float* __restrict__ xyz, int n, int npoint,
                      int* __restrict__ outIdx) {
  __shared__ float dist[8192];
  __shared__ float rmax[256];
  __shared__ int   rid[256];
  int b = blockIdx.x, t = threadIdx.x;
  const float* X = xyz + (size_t)b * n * 3;
  for (int i = t; i < n; i += 256) dist[i] = 1e10f;
  __syncthreads();
  int far = 0;
  for (int it = 0; it < npoint; ++it) {
    if (t == 0) outIdx[b * npoint + it] = far;
    float cx = X[far * 3 + 0], cy = X[far * 3 + 1], cz = X[far * 3 + 2];
    float best = -1.0f; int bi = n;
    for (int i = t; i < n; i += 256) {
      float dx = X[i * 3 + 0] - cx, dy = X[i * 3 + 1] - cy, dz = X[i * 3 + 2] - cz;
      float d = dx * dx + dy * dy + dz * dz;
      float m = dist[i]; m = d < m ? d : m; dist[i] = m;
      if (m > best) { best = m; bi = i; }
    }
    rmax[t] = best; rid[t] = bi;
    __syncthreads();
    for (int s = 128; s > 0; s >>= 1) {
      if (t < s) {
        if (rmax[t + s] > rmax[t] ||
            (rmax[t + s] == rmax[t] && rid[t + s] < rid[t])) {
          rmax[t] = rmax[t + s]; rid[t] = rid[t + s];
        }
      }
      __syncthreads();
    }
    far = rid[0];
    __syncthreads();
  }
}

// gather centroids: new_xyz[b][s] = xyz[b][idx[b][s]]
__global__ void k_gather_xyz(const float* __restrict__ xyz,
                             const int* __restrict__ idx,
                             float* __restrict__ out, int n, int np) {
  int i = blockIdx.x * 256 + threadIdx.x;
  if (i >= NBATCH * np) return;
  int b = i / np; int j = idx[i];
  for (int c = 0; c < 3; ++c)
    out[(size_t)i * 3 + c] = xyz[((size_t)b * n + j) * 3 + c];
}

// ---------------------------------------------------------------------------
// 2) ball query: first-k neighbors (index order) within radius, pad w/ first
// ---------------------------------------------------------------------------
__global__ void k_ballquery(const float* __restrict__ xyz,
                            const float* __restrict__ newXyz,
                            int* __restrict__ ballIdx,
                            int n, int np, int k, float r2) {
  int i = blockIdx.x * 256 + threadIdx.x;
  if (i >= NBATCH * np) return;
  int b = i / np;
  const float* X = xyz + (size_t)b * n * 3;
  float cx = newXyz[i * 3 + 0], cy = newXyz[i * 3 + 1], cz = newXyz[i * 3 + 2];
  int* o = ballIdx + (size_t)i * k;
  int cnt = 0;
  for (int j = 0; j < n && cnt < k; ++j) {
    float dx = X[j * 3 + 0] - cx, dy = X[j * 3 + 1] - cy, dz = X[j * 3 + 2] - cz;
    if (dx * dx + dy * dy + dz * dz <= r2) o[cnt++] = j;
  }
  int f = (cnt > 0) ? o[0] : 0;
  for (; cnt < k; ++cnt) o[cnt] = f;
}

// build grouped GEMM input rows: [feat(j), xyz(j)-center], zero-pad to kpad
__global__ void k_group(const int* __restrict__ ballIdx,
                        const float* __restrict__ feat,
                        const float* __restrict__ xyz,
                        const float* __restrict__ newXyz,
                        h16* __restrict__ G,
                        int n, int np, int k, int cfeat, int kpad) {
  long r = (long)blockIdx.x * 256 + threadIdx.x;
  long total = (long)NBATCH * np * k;
  if (r >= total) return;
  int b = (int)(r / ((long)np * k));
  int s = (int)((r / k) % np);
  int j = ballIdx[r];
  const float* F = feat + ((size_t)b * n + j) * cfeat;
  const float* P = xyz + ((size_t)b * n + j) * 3;
  const float* C = newXyz + ((size_t)b * np + s) * 3;
  h16* o = G + (size_t)r * kpad;
  int c = 0;
  for (; c < cfeat; ++c) o[c] = (h16)F[c];
  for (int q = 0; q < 3; ++q) { o[c] = (h16)(P[q] - C[q]); ++c; }
  for (; c < kpad; ++c) o[c] = (h16)0.f;
}

// ---------------------------------------------------------------------------
// 3) fold BN into weights, swizzled K32-chunked N-major layout:
//    Wp[((ks*npad + n)*32) + kk] = W[n][ks*32+kk] * s * gm[n]
//    bp[n] = bias[n]*s*gm[n] + bt[n]
// ---------------------------------------------------------------------------
__global__ void k_prep_w(const float* __restrict__ W, const float* __restrict__ bias,
                         const float* __restrict__ gm, const float* __restrict__ bt,
                         h16* __restrict__ Wp, float* __restrict__ bp,
                         int cin, int cout, int kpad, int npad) {
  int i = blockIdx.x * 256 + threadIdx.x;
  if (i < kpad * npad) {
    int kk = i & 31;
    int rest = i >> 5;
    int nn = rest % npad;
    int ks = rest / npad;
    int k = ks * 32 + kk;
    float v = 0.f;
    if (k < cin && nn < cout) v = W[(size_t)nn * cin + k] * (BN_SCALE_F * gm[nn]);
    Wp[i] = (h16)v;
  }
  if (i < npad) bp[i] = (i < cout) ? bias[i] * (BN_SCALE_F * gm[i]) + bt[i] : 0.f;
}

// f32 [B][K][N] -> swizzled f16 B-matrix (K32-chunked, N-major), per batch
__global__ void k_swz_b(const float* __restrict__ src, h16* __restrict__ dst,
                        int K, int N) {
  long i = (long)blockIdx.x * 256 + threadIdx.x;
  long per = (long)K * N;
  if (i >= (long)NBATCH * per) return;
  int b = (int)(i / per);
  long r = i % per;
  int kk = (int)(r & 31);
  long rest = r >> 5;
  int nn = (int)(rest % N);
  int ks = (int)(rest / N);
  int k = ks * 32 + kk;
  dst[(size_t)b * per + r] = (h16)src[(size_t)b * per + (size_t)k * N + nn];
}

// f32 -> padded f16 rows (row-major A operand)
__global__ void k_cvt_f16(const float* __restrict__ src, h16* __restrict__ dst,
                          long rows, int cin, int kpad) {
  long i = (long)blockIdx.x * 256 + threadIdx.x;
  if (i >= rows * kpad) return;
  long r = i / kpad; int c = (int)(i % kpad);
  dst[i] = (c < cin) ? (h16)src[r * cin + c] : (h16)0.f;
}

// ---------------------------------------------------------------------------
// 4) WMMA GEMM: out = act( A[M x Kpad] * B[Kpad x Npad] + bias )
//    A row-major; B pre-swizzled K32-chunked N-major. Block tile 128x64,
//    8 waves (4x2), wave tile 32x32 = 2x2 WMMA, double-buffered LDS.
// ---------------------------------------------------------------------------
__global__ void __launch_bounds__(256)
k_wmma_gemm(const h16* __restrict__ A, const h16* __restrict__ Bw,
            const float* __restrict__ bias,
            h16* __restrict__ outH, float* __restrict__ outF,
            int M, int Nexact, int Nstore, int Kpad, int Npad,
            long sA, long sB, long sOut, int relu) {
  __shared__ h16 As[2][128][32];   // [m][k]
  __shared__ h16 Bs[2][64][32];    // [n][k]  (swizzled: lane fragment contiguous)

  const h16* Ab = A + (size_t)blockIdx.z * sA;
  const h16* Bb = Bw + (size_t)blockIdx.z * sB;

  int t = threadIdx.x;
  int m0 = blockIdx.y * 128, n0 = blockIdx.x * 64;
  int wave = t >> 5, lane = t & 31, half = lane >> 4, l16 = lane & 15;
  int wm = wave & 3, wn = wave >> 2;

  int aRow = t >> 1, aCol = (t & 1) * 16;       // 16 halves per thread (A)

  auto loadA = [&](int buf, int ks) {
    V4 v0 = {}, v1 = {};
    int gr = m0 + aRow;
    if (gr < M) {
      const V4* s = (const V4*)(Ab + (size_t)gr * Kpad + (size_t)ks * 32 + aCol);
      v0 = s[0]; v1 = s[1];
    }
    *(V4*)&As[buf][aRow][aCol] = v0;
    *(V4*)&As[buf][aRow][aCol + 8] = v1;
  };
  // B tile for (ks, n0) is one contiguous 4KB block in the swizzled layout
  auto loadB = [&](int buf, int ks) {
    const V4* s = (const V4*)(Bb + ((size_t)ks * Npad + n0) * 32);
    ((V4*)&Bs[buf][0][0])[t] = s[t];            // 256 threads x 16B = 4KB
  };

  v8f acc[2][2] = {};
  int nK = Kpad >> 5;

  loadA(0, 0); loadB(0, 0);
  __syncthreads();

  for (int ks = 0; ks < nK; ++ks) {
    int cur = ks & 1;
    if (ks + 1 < nK) {
      loadA(cur ^ 1, ks + 1);
      loadB(cur ^ 1, ks + 1);
      if (ks + 2 < nK && (m0 + aRow) < M)
        __builtin_prefetch(Ab + (size_t)(m0 + aRow) * Kpad + (size_t)(ks + 2) * 32, 0, 1);
    }

    v16h af[2], bf[2];
#pragma unroll
    for (int mi = 0; mi < 2; ++mi) {
      // A 16x32 fragment: lane(l16) = row M, one contiguous 16-half
      // K chunk at [half*16, +16) (matches B fragment order)
      const h16* ar = &As[cur][wm * 32 + mi * 16 + l16][0];
#pragma unroll
      for (int i = 0; i < 8; ++i) {
        int k0 = half * 16 + i * 2;
        af[mi][2 * i]     = ar[k0];
        af[mi][2 * i + 1] = ar[k0 + 1];
      }
    }
#pragma unroll
    for (int ni = 0; ni < 2; ++ni) {
      // B 32x16 fragment (this code's contiguous k order, same as A): VGPR j holds K = half*16 + 2j (+1),
      // N = l16 -> 16 contiguous halves in the n-major swizzled tile
      const h16* br = &Bs[cur][wn * 32 + ni * 16 + l16][half * 16];
#pragma unroll
      for (int q = 0; q < 16; ++q) bf[ni][q] = br[q];
    }
#pragma unroll
    for (int mi = 0; mi < 2; ++mi)
#pragma unroll
      for (int ni = 0; ni < 2; ++ni)
        acc[mi][ni] = __builtin_amdgcn_wmma_f32_16x16x32_f16(
            false, af[mi], false, bf[ni], (short)0, acc[mi][ni], false, false);
    __syncthreads();
  }

  // epilogue: C/D layout — VGPR r: lanes 0-15 -> M=r, lanes 16-31 -> M=8+r
#pragma unroll
  for (int mi = 0; mi < 2; ++mi) {
#pragma unroll
    for (int ni = 0; ni < 2; ++ni) {
      int gn = n0 + wn * 32 + ni * 16 + l16;
#pragma unroll
      for (int r = 0; r < 8; ++r) {
        int gm_ = m0 + wm * 32 + mi * 16 + half * 8 + r;
        if (gm_ < M && gn < Nstore) {
          float v = 0.f;
          if (gn < Nexact) {
            v = acc[mi][ni][r];
            if (bias) v += bias[gn];
            if (relu) v = fmaxf(v, 0.f);
          }
          if (outH)
            outH[(size_t)blockIdx.z * sOut + (size_t)gm_ * Nstore + gn] = (h16)v;
          if (outF && gn < Nexact)
            outF[(size_t)blockIdx.z * sOut + (size_t)gm_ * Nexact + gn] = v;
        }
      }
    }
  }
}

// ---------------------------------------------------------------------------
// 5) neighbor max-pool: pooled[b][s][coff+c] = max_j act[(b,s,j)][c]
// ---------------------------------------------------------------------------
__global__ void k_maxpool(const h16* __restrict__ act, float* __restrict__ pooled,
                          int np, int k, int cout, int w, int ctot, int coff) {
  long i = (long)blockIdx.x * 256 + threadIdx.x;
  long total = (long)NBATCH * np * cout;
  if (i >= total) return;
  int c = (int)(i % cout);
  long s = i / cout;                     // b*np + s
  const h16* a = act + (size_t)s * k * w + c;
  float m = -1e30f;
  for (int j = 0; j < k; ++j) {
    float v = (float)a[(size_t)j * w];
    m = v > m ? v : m;
  }
  pooled[s * (size_t)ctot + coff + c] = m;
}

// ---------------------------------------------------------------------------
// 6) softmax over "points": out[b][kk][p] = softmax_p logits[(b*P+p)][kk]
// ---------------------------------------------------------------------------
__global__ void k_softmax_t(const float* __restrict__ logits,
                            float* __restrict__ out, int P, int K) {
  __shared__ float red[256];
  int b = blockIdx.x / K, kk = blockIdx.x % K, t = threadIdx.x;
  const float* L = logits + (size_t)b * P * K + kk;
  float mx = -1e30f;
  for (int p = t; p < P; p += 256) mx = fmaxf(mx, L[(size_t)p * K]);
  red[t] = mx; __syncthreads();
  for (int s = 128; s; s >>= 1) { if (t < s) red[t] = fmaxf(red[t], red[t + s]); __syncthreads(); }
  mx = red[0]; __syncthreads();
  float sm = 0.f;
  for (int p = t; p < P; p += 256) sm += __expf(L[(size_t)p * K] - mx);
  red[t] = sm; __syncthreads();
  for (int s = 128; s; s >>= 1) { if (t < s) red[t] += red[t + s]; __syncthreads(); }
  float inv = 1.f / red[0];
  float* O = out + ((size_t)b * K + kk) * P;
  for (int p = t; p < P; p += 256) O[p] = __expf(L[(size_t)p * K] - mx) * inv;
}

// ---------------------------------------------------------------------------
// 7) small einsums
// ---------------------------------------------------------------------------
__global__ void k_apply_w3(const float* __restrict__ w, const float* __restrict__ pts,
                           float* __restrict__ out, int K, int P) {
  int i = blockIdx.x * 256 + threadIdx.x;
  if (i >= NBATCH * K * 3) return;
  int c = i % 3, kk = (i / 3) % K, b = i / (3 * K);
  const float* W = w + ((size_t)b * K + kk) * P;
  const float* X = pts + (size_t)b * P * 3 + c;
  float s = 0.f;
  for (int p = 0; p < P; ++p) s += W[p] * X[(size_t)p * 3];
  out[i] = s;
}

__global__ void k_mindist(const float* __restrict__ l3, const float* __restrict__ skel,
                          float* __restrict__ md, int P, int K) {
  int i = blockIdx.x * 256 + threadIdx.x;
  if (i >= NBATCH * P) return;
  int b = i / P;
  const float* x = l3 + (size_t)i * 3;
  float best = 1e30f;
  for (int kk = 0; kk < K; ++kk) {
    const float* s = skel + ((size_t)b * K + kk) * 3;
    float dx = x[0] - s[0], dy = x[1] - s[1], dz = x[2] - s[2];
    best = fminf(best, sqrtf(dx * dx + dy * dy + dz * dz));
  }
  md[i] = best;
}

__global__ void k_skelr(const float* __restrict__ w, const float* __restrict__ md,
                        float* __restrict__ out, int K, int P) {
  int i = blockIdx.x * 256 + threadIdx.x;
  if (i >= NBATCH * K) return;
  int b = i / K;
  const float* W = w + (size_t)i * P;
  const float* M = md + (size_t)b * P;
  float s = 0.f;
  for (int p = 0; p < P; ++p) s += W[p] * M[p];
  out[i] = s;
}

__global__ void k_nori(const float* __restrict__ norw, const float* __restrict__ skel,
                       float* __restrict__ out, int K) {
  int i = blockIdx.x * 256 + threadIdx.x;
  if (i >= NBATCH * K * 3) return;
  int c = i % 3, kk = (i / 3) % K, b = i / (3 * K);
  const float* W = norw + ((size_t)b * K + kk) * K;
  float s = 0.f;
  for (int p = 0; p < K; ++p) s += W[p] * skel[((size_t)b * K + p) * 3 + c];
  out[i] = s - skel[((size_t)b * K + kk) * 3 + c];
}

__global__ void k_copy(const float* __restrict__ s, float* __restrict__ d, long n) {
  long i = (long)blockIdx.x * 256 + threadIdx.x;
  if (i < n) d[i] = s[i];
}

// ---------------------------------------------------------------------------
// host orchestration
// ---------------------------------------------------------------------------
#define NBLK(total) ((unsigned)(((long)(total) + 255) / 256))

extern "C" void kernel_launch(void* const* d_in, const int* in_sizes, int n_in,
                              void* d_out, int out_size, void* d_ws, size_t ws_size,
                              hipStream_t stream) {
  (void)in_sizes; (void)n_in; (void)out_size; (void)ws_size;

  const int N0 = 8192;
  static const int   NP[4]     = {1024, 768, 512, 512};
  static const float RAD[4][2] = {{0.1f, 0.2f}, {0.2f, 0.4f}, {0.4f, 0.6f}, {0.6f, 0.8f}};
  static const int   KS[4][2]  = {{16, 32}, {32, 64}, {32, 64}, {64, 128}};
  static const int   CH[4][4]  = {{6, 16, 16, 32}, {67, 32, 32, 64},
                                  {131, 64, 64, 128}, {259, 128, 128, 256}};
  static const int   HCH[6]    = {512, 384, 256, 256, 128, 100};

  // ---- workspace bump allocator -------------------------------------------
  char* wsb = (char*)d_ws;
  size_t off = 0;
  auto alloc = [&](size_t bytes) -> void* {
    void* p = wsb + off;
    off = (off + bytes + 255) & ~(size_t)255;
    return p;
  };
  float* lxyz0   = (float*)alloc((size_t)NBATCH * N0 * 3 * 4);
  float* feat0   = (float*)alloc((size_t)NBATCH * N0 * 3 * 4);
  int*   fpsIdx  = (int*)  alloc((size_t)NBATCH * 1024 * 4);
  float* nx0     = (float*)alloc((size_t)NBATCH * 1024 * 3 * 4);
  float* nx1     = (float*)alloc((size_t)NBATCH * 1024 * 3 * 4);
  int*   ballIdx = (int*)  alloc((size_t)NBATCH * 512 * 128 * 4);
  h16*   G       = (h16*)  alloc((size_t)262144 * 288 * 2);
  h16*   actA    = (h16*)  alloc((size_t)262144 * 256 * 2);
  h16*   actB    = (h16*)  alloc((size_t)262144 * 256 * 2);
  float* pooled0 = (float*)alloc((size_t)NBATCH * 1024 * 512 * 4);
  float* pooled1 = (float*)alloc((size_t)NBATCH * 1024 * 512 * 4);
  h16*   wbuf    = (h16*)  alloc((size_t)512 * 448 * 2);
  float* bbuf    = (float*)alloc((size_t)448 * 4);
  h16*   ctxF16  = (h16*)  alloc((size_t)2048 * 512 * 2);
  h16*   ctxSwz  = (h16*)  alloc((size_t)2048 * 512 * 2);
  h16*   hact0   = (h16*)  alloc((size_t)2048 * 384 * 2);
  h16*   hact1   = (h16*)  alloc((size_t)2048 * 384 * 2);
  float* logits  = (float*)alloc((size_t)2048 * 100 * 4);
  float* minD    = (float*)alloc((size_t)2048 * 4);
  h16*   wF16    = (h16*)  alloc((size_t)400 * 512 * 2);
  h16*   scF16   = (h16*)  alloc((size_t)400 * 512 * 2);
  float* logits2 = (float*)alloc((size_t)400 * 100 * 4);
  float* norW    = (float*)alloc((size_t)NBATCH * 100 * 100 * 4);

  // ---- d_out regions (tuple order) ----------------------------------------
  float* out    = (float*)d_out;
  float* o_skel = out;               // (4,100,3)   1200
  float* o_r    = out + 1200;        // (4,100,1)   400
  float* o_cmb  = out + 1600;        // (4,100,512) 204800
  float* o_nori = out + 206400;      // (4,100,3)   1200
  float* o_w    = out + 207600;      // (4,100,512) 204800
  float* o_l3a  = out + 412400;      // (4,512,3)   6144
  float* o_l3b  = out + 418544;      // (4,512,3)   6144

  // ---- parameter walker (dict order: xyz, sa, head, head_nor) -------------
  int pi = 1;
  auto nextLin = [&](const float** W, const float** b, const float** g, const float** t) {
    *W = (const float*)d_in[pi + 0];
    *b = (const float*)d_in[pi + 1];
    *g = (const float*)d_in[pi + 2];
    *t = (const float*)d_in[pi + 3];
    pi += 4;
  };

  auto gemmLayer = [&](const h16* Ain, int rows, int kpad,
                       const float* W, const float* bi, const float* g, const float* t,
                       int cin, int cout, int relu, h16* outH, float* outF) {
    int npad = al64(cout), nstore = al32(cout);
    k_prep_w<<<NBLK(kpad * npad), 256, 0, stream>>>(W, bi, g, t, wbuf, bbuf,
                                                    cin, cout, kpad, npad);
    dim3 grid((nstore + 63) / 64, (rows + 127) / 128, 1);
    k_wmma_gemm<<<grid, 256, 0, stream>>>(Ain, wbuf, bbuf, outH, outF,
                                          rows, cout, nstore, kpad, npad,
                                          0L, 0L, 0L, relu);
    return nstore;
  };

  // ---- stage 0: split input ------------------------------------------------
  const float* xyz6 = (const float*)d_in[0];
  k_transpose_in<<<NBLK(NBATCH * N0), 256, 0, stream>>>(xyz6, lxyz0, feat0, N0);

  // ---- SA levels -----------------------------------------------------------
  float* curXyz = lxyz0;
  int    curN   = N0;
  float* nxb[2] = {nx0, nx1};
  float* pl[2]  = {pooled0, pooled1};

  for (int lvl = 0; lvl < 4; ++lvl) {
    int np = NP[lvl];
    k_fps<<<NBATCH, 256, 0, stream>>>(curXyz, curN, np, fpsIdx);
    float* nxyz = nxb[lvl & 1];
    k_gather_xyz<<<NBLK(NBATCH * np), 256, 0, stream>>>(curXyz, fpsIdx, nxyz, curN, np);

    int cfeat = CH[lvl][0] - 3;
    const float* feats = (lvl == 0) ? feat0 : pl[(lvl + 1) & 1];
    float* pooledOut = pl[lvl & 1];
    int ctot = 2 * CH[lvl][3];

    for (int sc = 0; sc < 2; ++sc) {
      int k = KS[lvl][sc];
      float r = RAD[lvl][sc];
      k_ballquery<<<NBLK(NBATCH * np), 256, 0, stream>>>(curXyz, nxyz, ballIdx,
                                                         curN, np, k, r * r);
      int kpad0 = al32(CH[lvl][0]);
      long rows = (long)NBATCH * np * k;
      k_group<<<NBLK(rows), 256, 0, stream>>>(ballIdx, feats, curXyz, nxyz, G,
                                              curN, np, k, cfeat, kpad0);
      const h16* Ain = G;
      int kpad = kpad0;
      h16* bufs[2] = {actA, actB};
      h16* last = nullptr;
      for (int l = 0; l < 3; ++l) {
        const float *W, *bi, *g, *t;
        nextLin(&W, &bi, &g, &t);
        h16* Aout = bufs[l & 1];
        kpad = gemmLayer(Ain, (int)rows, kpad, W, bi, g, t,
                         CH[lvl][l], CH[lvl][l + 1], 1, Aout, nullptr);
        Ain = Aout; last = Aout;
      }
      k_maxpool<<<NBLK((long)NBATCH * np * CH[lvl][3]), 256, 0, stream>>>(
          last, pooledOut, np, k, CH[lvl][3], kpad, ctot, sc * CH[lvl][3]);
    }
    curXyz = nxyz;
    curN = np;
  }

  float* l3xyz   = curXyz;   // (4,512,3)
  float* context = pl[1];    // (4,512,512) f32

  // ---- head: context -> weights -------------------------------------------
  {
    long rows = (long)NBATCH * 512;
    k_cvt_f16<<<NBLK(rows * 512), 256, 0, stream>>>(context, ctxF16, rows, 512, 512);
    const h16* Ain = ctxF16;
    int kpad = 512;
    h16* bufs[2] = {hact0, hact1};
    for (int l = 0; l < 5; ++l) {
      const float *W, *bi, *g, *t;
      nextLin(&W, &bi, &g, &t);
      if (l < 4) {
        h16* Aout = bufs[l & 1];
        kpad = gemmLayer(Ain, (int)rows, kpad, W, bi, g, t, HCH[l], HCH[l + 1], 1,
                         Aout, nullptr);
        Ain = Aout;
      } else {
        gemmLayer(Ain, (int)rows, kpad, W, bi, g, t, HCH[l], HCH[l + 1], 0,
                  nullptr, logits);
      }
    }
    k_softmax_t<<<NBATCH * 100, 256, 0, stream>>>(logits, o_w, 512, 100);
  }

  // ---- skeleton points, radii ---------------------------------------------
  k_apply_w3<<<NBLK(NBATCH * 100 * 3), 256, 0, stream>>>(o_w, l3xyz, o_skel, 100, 512);
  k_mindist<<<NBLK(NBATCH * 512), 256, 0, stream>>>(l3xyz, o_skel, minD, 512, 100);
  k_skelr<<<NBLK(NBATCH * 100), 256, 0, stream>>>(o_w, minD, o_r, 100, 512);

  // ---- shape_cmb = weights @ context (batched WMMA, B varies per batch) ----
  k_cvt_f16<<<NBLK((long)NBATCH * 100 * 512), 256, 0, stream>>>(o_w, wF16,
                                                                (long)NBATCH * 100, 512, 512);
  k_swz_b<<<NBLK((long)NBATCH * 512 * 512), 256, 0, stream>>>(context, ctxSwz, 512, 512);
  {
    dim3 grid(512 / 64, (100 + 127) / 128, NBATCH);
    k_wmma_gemm<<<grid, 256, 0, stream>>>(wF16, ctxSwz, nullptr, nullptr, o_cmb,
                                          100, 512, 512, 512, 512,
                                          (long)100 * 512, (long)512 * 512,
                                          (long)100 * 512, 0);
  }

  // ---- head_nor: shape_cmb -> nor_w ---------------------------------------
  {
    long rows = (long)NBATCH * 100;
    k_cvt_f16<<<NBLK(rows * 512), 256, 0, stream>>>(o_cmb, scF16, rows, 512, 512);
    const h16* Ain = scF16;
    int kpad = 512;
    h16* bufs[2] = {hact0, hact1};
    for (int l = 0; l < 5; ++l) {
      const float *W, *bi, *g, *t;
      nextLin(&W, &bi, &g, &t);
      if (l < 4) {
        h16* Aout = bufs[l & 1];
        kpad = gemmLayer(Ain, (int)rows, kpad, W, bi, g, t, HCH[l], HCH[l + 1], 1,
                         Aout, nullptr);
        Ain = Aout;
      } else {
        gemmLayer(Ain, (int)rows, kpad, W, bi, g, t, HCH[l], HCH[l + 1], 0,
                  nullptr, logits2);
      }
    }
    k_softmax_t<<<NBATCH * 100, 256, 0, stream>>>(logits2, norW, 100, 100);
  }

  // ---- skel_nori = nor_w @ skel_xyz - skel_xyz ----------------------------
  k_nori<<<NBLK(NBATCH * 100 * 3), 256, 0, stream>>>(norW, o_skel, o_nori, 100);

  // ---- l3_xyz copies -------------------------------------------------------
  k_copy<<<NBLK(NBATCH * 512 * 3), 256, 0, stream>>>(l3xyz, o_l3a, (long)NBATCH * 512 * 3);
  k_copy<<<NBLK(NBATCH * 512 * 3), 256, 0, stream>>>(l3xyz, o_l3b, (long)NBATCH * 512 * 3);
}
